// LSTMmodule_43817256354349
// MI455X (gfx1250) — matrix operand order repaired and checked by reading, not compiled, not run on hardware
//
#include <hip/hip_runtime.h>
#include <hip/hip_bf16.h>

// ---------------------------------------------------------------------------
// LSTM forward, MI455X / gfx1250.
//   - fused weights packed to bf16 [2048 cols][768 K], K-contiguous
//   - x transposed/converted to [T][B][I] bf16
//   - persistent kernel: 128 blocks x 256 thr (8 waves), one
//     (16 rows x 64 cols x 4 gates) macro tile per block.
//     K=768 streamed in 32-wide chunks, staged into double-buffered LDS via
//     global_load_async_to_lds_b{32,128} (ASYNCcnt), consumed by
//     v_wmma_f32_16x16x32_bf16, fused sigmoid/tanh + c/h update,
//     device-wide barrier per timestep.
// ---------------------------------------------------------------------------

typedef __attribute__((ext_vector_type(16))) __bf16 bf16x16;
typedef __attribute__((ext_vector_type(8)))  float  f32x8;

union Frag16 { bf16x16 v; uint4 q[2]; };

#define SEQ_T   512
#define BATCH_N 256
#define IN_I    256
#define HID_H   512
#define KDIM    768          // I + H
#define NCHUNK  (KDIM / 32)  // 24
#define NBLK    128          // persistent grid
#define NTHR    256          // 8 waves

// LDS layout (bytes): buf0 A[0,1024) B[1024,17408) ; buf1 A[17408,18432) B[18432,34816)
// G_sh overlaps buf0 [0,16384) after the K loop.
#define BUF_STRIDE 17408
#define SMEM_BYTES 34816

__device__ __forceinline__ unsigned short f2bf(float f) {
  unsigned u = __float_as_uint(f);
  unsigned r = (u + 0x7FFFu + ((u >> 16) & 1u)) >> 16;   // RNE
  return (unsigned short)r;
}

__device__ __forceinline__ float sigf(float x) {
  return 1.0f / (1.0f + __expf(-x));
}

// ---- CDNA5 async global->LDS copies (ASYNCcnt tracked) ---------------------
__device__ __forceinline__ void async_b32(unsigned lds, unsigned long long ga) {
  asm volatile("global_load_async_to_lds_b32 %0, %1, off"
               :: "v"(lds), "v"(ga) : "memory");
}
template <int OFF>
__device__ __forceinline__ void async_b128(unsigned lds, unsigned long long ga) {
  asm volatile("global_load_async_to_lds_b128 %0, %1, off offset:%2"
               :: "v"(lds), "v"(ga), "i"(OFF) : "memory");
}
__device__ __forceinline__ void wait_async_le5() {
  asm volatile("s_wait_asynccnt 5" ::: "memory");
}
__device__ __forceinline__ void wait_async_0() {
  asm volatile("s_wait_asynccnt 0" ::: "memory");
}

// ---------------- init: zero barrier counter, c state, both h buffers -------
__global__ void lstm_init(int* bar, unsigned int* c_u32, unsigned int* h_u32) {
  unsigned gid = blockIdx.x * blockDim.x + threadIdx.x;
  if (gid == 0) *bar = 0;
  if (gid < BATCH_N * HID_H)      c_u32[gid] = 0u;   // 131072 f32 words
  if (gid < BATCH_N * HID_H)      h_u32[gid] = 0u;   // 131072 u32 = 2 bf16 bufs
}

// ---------------- pack weights: Wpack[col][k] bf16, col in [0,2048) ---------
__global__ void pack_weights(const float* __restrict__ Wui, const float* __restrict__ Whi,
                             const float* __restrict__ Wuf, const float* __restrict__ Whf,
                             const float* __restrict__ Wug, const float* __restrict__ Whg,
                             const float* __restrict__ Wuo, const float* __restrict__ Who,
                             unsigned short* __restrict__ Wpack) {
  unsigned gid = blockIdx.x * blockDim.x + threadIdx.x;
  if (gid >= 2048u * KDIM) return;
  int k   = (int)(gid % KDIM);
  int col = (int)(gid / KDIM);
  int g   = col >> 9;          // gate 0..3  (i,f,g,o)
  int h   = col & 511;
  const float* Wu = (g == 0) ? Wui : (g == 1) ? Wuf : (g == 2) ? Wug : Wuo;
  const float* Wh = (g == 0) ? Whi : (g == 1) ? Whf : (g == 2) ? Whg : Who;
  float v = (k < IN_I) ? Wu[(size_t)k * HID_H + h]
                       : Wh[(size_t)(k - IN_I) * HID_H + h];
  Wpack[gid] = f2bf(v);
}

// ---------------- convert x: [B,T,I] f32 -> [T,B,I] bf16 --------------------
__global__ void convert_x(const float* __restrict__ x, unsigned int* __restrict__ xbf_u32) {
  unsigned gid = blockIdx.x * blockDim.x + threadIdx.x;     // one per 4 elems
  const unsigned total4 = (unsigned)(SEQ_T * BATCH_N * (IN_I / 4));
  if (gid >= total4) return;
  int i0 = (int)(gid % (IN_I / 4)) * 4;
  unsigned rest = gid / (IN_I / 4);
  int t = (int)(rest % SEQ_T);
  int b = (int)(rest / SEQ_T);
  const float4 v = *(const float4*)(x + ((size_t)b * SEQ_T + t) * IN_I + i0);
  unsigned short h0 = f2bf(v.x), h1 = f2bf(v.y), h2 = f2bf(v.z), h3 = f2bf(v.w);
  size_t oe = ((size_t)t * BATCH_N + b) * IN_I + i0;        // bf16 element idx
  uint2 o; o.x = (unsigned)h0 | ((unsigned)h1 << 16);
           o.y = (unsigned)h2 | ((unsigned)h3 << 16);
  *(uint2*)(xbf_u32 + oe / 2) = o;
}

// ---------------- persistent LSTM kernel ------------------------------------
__global__ __launch_bounds__(NTHR, 1) void lstm_persistent(
    const unsigned short* __restrict__ xbf,    // [T][B][I] bf16
    const unsigned short* __restrict__ Wpack,  // [2048][768] bf16
    const float* __restrict__ bi, const float* __restrict__ bff,
    const float* __restrict__ bg, const float* __restrict__ bo,
    float* __restrict__ c_state,               // [B][H] f32
    unsigned short* __restrict__ h_bf,         // [2][B][H] bf16 (ping-pong)
    int* __restrict__ bar,
    float* __restrict__ out)                   // [B][H] f32
{
  __shared__ __align__(16) unsigned char smem[SMEM_BYTES];
  const unsigned sb = (unsigned)(unsigned long long)(uintptr_t)smem; // LDS base

  const int tid  = threadIdx.x;
  const int lane = tid & 31;
  const int wave = tid >> 5;                  // 0..7
  const int m0   = (blockIdx.x >> 3) * 16;    // batch row base (16 tiles)
  const int n0   = (blockIdx.x & 7) * 64;     // hidden col base (8 groups)

  // per-thread staging geometry (constant across chunks)
  const int ar   = tid >> 4;                  // A row 0..15
  const int akw  = tid & 15;                  // A u32 slot in row
  const int bc   = tid;                       // B local col 0..255
  const int bg_  = bc >> 6;
  const int colg = bg_ * HID_H + n0 + (bc & 63);
  const unsigned long long wrow =
      (unsigned long long)(uintptr_t)(Wpack + (size_t)colg * KDIM);

  for (int t = 0; t < SEQ_T; ++t) {
    const unsigned short* hcur  = h_bf + (size_t)(t & 1) * (BATCH_N * HID_H);
    unsigned short*       hnext = h_bf + (size_t)((t + 1) & 1) * (BATCH_N * HID_H);

    f32x8 acc0 = {};
    f32x8 acc1 = {};

    // issue chunk 0 into buf0 (chunk 0 reads x only)
    {
      const unsigned base = sb;               // buf0
      int k = akw * 2;                        // kc = 0
      async_b32(base + tid * 4,
                (unsigned long long)(uintptr_t)
                    (xbf + ((size_t)t * BATCH_N + (m0 + ar)) * IN_I + k));
      unsigned ldsb = base + 1024 + bc * 64;
      unsigned long long gb = wrow;           // kc*64 bytes = 0
      async_b128<0 >(ldsb, gb);
      async_b128<16>(ldsb, gb);
      async_b128<32>(ldsb, gb);
      async_b128<48>(ldsb, gb);
    }

    for (int kc = 0; kc < NCHUNK; ++kc) {
      const int buf = kc & 1;
      const unsigned cbase = sb + (unsigned)buf * BUF_STRIDE;        // consume
      const unsigned nbase = sb + (unsigned)(buf ^ 1) * BUF_STRIDE;  // fill

      // ---- issue next chunk's asyncs into the other buffer ----
      if (kc + 1 < NCHUNK) {
        int kn = kc + 1;
        int k  = kn * 32 + akw * 2;
        unsigned long long ga;
        if (k < IN_I)
          ga = (unsigned long long)(uintptr_t)
                   (xbf + ((size_t)t * BATCH_N + (m0 + ar)) * IN_I + k);
        else
          ga = (unsigned long long)(uintptr_t)
                   (hcur + (size_t)(m0 + ar) * HID_H + (k - IN_I));
        async_b32(nbase + tid * 4, ga);
        unsigned ldsb = nbase + 1024 + bc * 64;
        unsigned long long gb = wrow + (unsigned long long)kn * 64;
        async_b128<0 >(ldsb, gb);
        async_b128<16>(ldsb, gb);
        async_b128<32>(ldsb, gb);
        async_b128<48>(ldsb, gb);
        wait_async_le5();                     // current chunk's 5 retired
      } else {
        wait_async_0();
      }
      __syncthreads();                        // all waves' current chunk in LDS

      // ---- fragments from LDS (ISA 16-bit A 16x32 / B 32x16 layouts) ----
      const unsigned short* A_sh = (const unsigned short*)(smem + buf * BUF_STRIDE);
      const unsigned short* B_sh = (const unsigned short*)(smem + buf * BUF_STRIDE + 1024);
      Frag16 af;
      {
        int r  = lane & 15;
        int kk = (lane < 16) ? 0 : 8;
        af.q[0] = *(const uint4*)(A_sh + r * 32 + kk);        // K = kk..kk+7
        af.q[1] = *(const uint4*)(A_sh + r * 32 + kk + 16);   // K = kk+16..kk+23
      }
#pragma unroll
      for (int j = 0; j < 2; ++j) {
        int idx = wave * 2 + j;
        int g   = idx >> 2;
        int ct  = idx & 3;
        int c   = g * 64 + ct * 16 + (lane & 15);
        int ko  = (lane < 16) ? 0 : 8;
        Frag16 bfr;
        bfr.q[0] = *(const uint4*)(B_sh + c * 32 + ko);
        bfr.q[1] = *(const uint4*)(B_sh + c * 32 + ko + 16);
        f32x8& acc = j ? acc1 : acc0;
        acc = __builtin_amdgcn_wmma_f32_16x16x32_bf16(
                false, af.v, false, bfr.v, (short)0, acc, false, false);
      }
      __syncthreads();                        // frag reads done before refill
    }

    // ---- spill gate tiles to LDS: G_sh[m][gate][64] (overlaps buf0) ----
    float* G_sh = (float*)(smem);
#pragma unroll
    for (int j = 0; j < 2; ++j) {
      int idx   = wave * 2 + j;
      int g     = idx >> 2;
      int ct    = idx & 3;
      int n     = lane & 15;
      int mbase = (lane < 16) ? 0 : 8;
      f32x8 acc = j ? acc1 : acc0;
#pragma unroll
      for (int v = 0; v < 8; ++v) {
        G_sh[((mbase + v) * 4 + g) * 64 + ct * 16 + n] = acc[v];
      }
    }
    __syncthreads();

    // ---- fused gate math + state update: 1024 elems / 256 threads ----
#pragma unroll
    for (int e = 0; e < 4; ++e) {
      int el = tid * 4 + e;
      int r  = el >> 6;                       // 0..15
      int cc = el & 63;
      int ch = n0 + cc;
      float gi = G_sh[(r * 4 + 0) * 64 + cc] + bi [ch];
      float gf = G_sh[(r * 4 + 1) * 64 + cc] + bff[ch];
      float gg = G_sh[(r * 4 + 2) * 64 + cc] + bg [ch];
      float go = G_sh[(r * 4 + 3) * 64 + cc] + bo [ch];
      float it = sigf(gi);
      float ft = sigf(gf);
      float gt = tanhf(gg);
      float ot = sigf(go);
      size_t off = (size_t)(m0 + r) * HID_H + ch;
      float cn = ft * c_state[off] + it * gt;
      c_state[off] = cn;
      float hn = ot * tanhf(cn);
      hnext[off] = f2bf(hn);
      if (t == SEQ_T - 1) out[off] = hn;
    }

    // ---- device-wide barrier (monotonic counter, no reset) ----
    __threadfence();
    __syncthreads();
    if (tid == 0) {
      __hip_atomic_fetch_add(bar, 1, __ATOMIC_ACQ_REL, __HIP_MEMORY_SCOPE_AGENT);
      const int target = (t + 1) * NBLK;
      while (__hip_atomic_load(bar, __ATOMIC_ACQUIRE, __HIP_MEMORY_SCOPE_AGENT) < target) {
        __builtin_amdgcn_s_sleep(2);
      }
    }
    __syncthreads();                          // also guards G_sh / buf0 reuse
  }
}

// ---------------------------------------------------------------------------
extern "C" void kernel_launch(void* const* d_in, const int* in_sizes, int n_in,
                              void* d_out, int out_size, void* d_ws, size_t ws_size,
                              hipStream_t stream) {
  const float* x    = (const float*)d_in[0];
  const float* W_ui = (const float*)d_in[1];
  const float* W_hi = (const float*)d_in[2];
  const float* b_i  = (const float*)d_in[3];
  const float* W_uf = (const float*)d_in[4];
  const float* W_hf = (const float*)d_in[5];
  const float* b_f  = (const float*)d_in[6];
  const float* W_ug = (const float*)d_in[7];
  const float* W_hg = (const float*)d_in[8];
  const float* b_g  = (const float*)d_in[9];
  const float* W_uo = (const float*)d_in[10];
  const float* W_ho = (const float*)d_in[11];
  const float* b_o  = (const float*)d_in[12];

  char* ws = (char*)d_ws;
  // workspace layout (bytes, 256-aligned)
  const size_t off_bar = 0;                                   // int counter
  const size_t off_c   = 256;                                 // 512 KB f32
  const size_t off_h   = off_c + (size_t)BATCH_N * HID_H * 4; // 2x256KB bf16
  const size_t off_w   = off_h + (size_t)2 * BATCH_N * HID_H * 2;
  const size_t off_x   = off_w + (size_t)2048 * KDIM * 2;     // 3 MB
  // xbf: 64 MB at off_x

  int*            bar   = (int*)(ws + off_bar);
  float*          c_st  = (float*)(ws + off_c);
  unsigned short* h_bf  = (unsigned short*)(ws + off_h);
  unsigned short* Wpack = (unsigned short*)(ws + off_w);
  unsigned short* xbf   = (unsigned short*)(ws + off_x);

  // 1) zero barrier + states
  lstm_init<<<(BATCH_N * HID_H + NTHR - 1) / NTHR, NTHR, 0, stream>>>(
      bar, (unsigned int*)c_st, (unsigned int*)h_bf);

  // 2) pack fused bf16 weights, K-contiguous per column
  {
    unsigned total = 2048u * KDIM;
    pack_weights<<<(total + NTHR - 1) / NTHR, NTHR, 0, stream>>>(
        W_ui, W_hi, W_uf, W_hf, W_ug, W_hg, W_uo, W_ho, Wpack);
  }

  // 3) x -> [T,B,I] bf16
  {
    unsigned total4 = (unsigned)(SEQ_T * BATCH_N * (IN_I / 4));
    convert_x<<<(total4 + NTHR - 1) / NTHR, NTHR, 0, stream>>>(x, (unsigned int*)xbf);
  }

  // 4) persistent recurrence
  lstm_persistent<<<NBLK, NTHR, 0, stream>>>(
      xbf, Wpack, b_i, b_f, b_g, b_o, c_st, h_bf, bar, (float*)d_out);
}
